// DynamicEdge_3367254360320
// MI455X (gfx1250) — matrix operand order repaired and checked by reading, not compiled, not run on hardware
//
#include <hip/hip_runtime.h>

// ---------------------------------------------------------------------------
// DGCNN (dynamic EdgeConv) for MI455X / gfx1250, wave32 + WMMA bf16.
// GEMM work (KNN Gram, EdgeConv MLPs, head MLP) -> v_wmma_f32_16x16x32_bf16.
// Distance matrix never touches HBM: WMMA tile -> distance -> wave-wide
// bitonic top-32 in registers (k == warpSize == 32), with a threshold
// early-out. Column tiles are staged LDS-side with async DMA
// (global_load_async_to_lds_b128) and double buffering.
// ---------------------------------------------------------------------------

#define DEVFN __device__ __forceinline__
#define AS1 __attribute__((address_space(1)))
#define AS3 __attribute__((address_space(3)))

#if __has_builtin(__builtin_amdgcn_global_load_async_to_lds_b128)
#define HAS_ASYNC_LDS 1
#else
#define HAS_ASYNC_LDS 0
#endif

typedef __attribute__((ext_vector_type(16))) __bf16 bfx16;
typedef __attribute__((ext_vector_type(8)))  float  f32x8;
typedef int v4i __attribute__((vector_size(16)));

constexpr int N_PTS = 16384;
constexpr int C0    = 64;
constexpr int H     = 128;
constexpr int KNN   = 32;
constexpr int NOUT  = 64;
constexpr float EPSV = 1e-5f;

DEVFN f32x8 wmma_bf16(bfx16 a, bfx16 b, f32x8 c) {
  return __builtin_amdgcn_wmma_f32_16x16x32_bf16(false, a, false, b, (short)0, c,
                                                 false, false);
}

template <int Nw>
DEVFN void wait_async() {
  asm volatile("s_wait_asynccnt %0" : : "n"(Nw) : "memory");
}

// one 16-byte async (or fallback sync) global->LDS copy
DEVFN void copy16_async(const void* g, void* l) {
#if HAS_ASYNC_LDS
  __builtin_amdgcn_global_load_async_to_lds_b128((AS1 v4i*)g, (AS3 v4i*)l, 0, 0);
#else
  *(uint4*)l = *(const uint4*)g;
#endif
}

// A-matrix 16x32 bf16 lane layout (ISA 7.12.2): lane = half*16 + m,
// elems [0..7] = K {half*8 ..}, elems [8..15] = K {16+half*8 ..}.
DEVFN bfx16 load_a(const __bf16* row, int kb, int half) {
  bfx16 a;
#pragma unroll
  for (int i = 0; i < 8; ++i) {
    a[i]     = row[kb + half * 8 + i];
    a[i + 8] = row[kb + 16 + half * 8 + i];
  }
  return a;
}

// B-matrix 32x16 bf16: same K order as A: elems [0..7] = K {half*8 ..},
// elems [8..15] = K {16+half*8 ..}; col = lane%16.
DEVFN bfx16 load_b(const __bf16* row, int kb, int half) {
  bfx16 b;
#pragma unroll
  for (int i = 0; i < 8; ++i) {
    b[i]     = row[kb + half * 8 + i];
    b[i + 8] = row[kb + 16 + half * 8 + i];
  }
  return b;
}

// ---- wave-wide top-32 selection (ascending in `curV` across lanes) --------

DEVFN void bitonic_sort32(float& v, int& i) {
  const int lane = threadIdx.x & 31;
#pragma unroll
  for (int k = 2; k <= 32; k <<= 1) {
#pragma unroll
    for (int j = k >> 1; j > 0; j >>= 1) {
      float ov = __shfl_xor(v, j, 32);
      int   oi = __shfl_xor(i, j, 32);
      bool keepMin = (((lane & j) == 0) == ((lane & k) == 0));
      if ((ov < v) == keepMin) { v = ov; i = oi; }
    }
  }
}

DEVFN void topk_merge(float& curV, int& curI, float nv, int ni) {
  const int lane = threadIdx.x & 31;
  bitonic_sort32(nv, ni);                     // batch ascending
  float rv = __shfl(nv, 31 - lane, 32);       // reversed -> descending
  int   ri = __shfl(ni, 31 - lane, 32);
  if (rv < curV) { curV = rv; curI = ri; }    // 32 smallest of 64, bitonic
#pragma unroll
  for (int j = 16; j > 0; j >>= 1) {          // bitonic cleanup -> ascending
    float ov = __shfl_xor(curV, j, 32);
    int   oi = __shfl_xor(curI, j, 32);
    bool keepMin = ((lane & j) == 0);
    if ((ov < curV) == keepMin) { curV = ov; curI = oi; }
  }
}

// ---- KNN: Gram-WMMA + distance + streaming top-32, async double-buffered --
// WG = 8 waves owns 32 rows; sweeps columns in super-tiles of 64.
// Tiles: 2 M-tiles x 4 N-tiles -> one 16x16 WMMA tile per wave.
template <int CIN>
__global__ __launch_bounds__(256)
void knn_kernel(const __bf16* __restrict__ xb, const float* __restrict__ sq,
                int* __restrict__ idx_out) {
  constexpr int SC = 64;            // columns per iteration
  constexpr int G  = CIN / 32;      // async b128 chunks per thread per group
  constexpr int GA = CIN / 64;      // chunks per thread for the A tile
  __shared__ __align__(16) __bf16 Atile[32 * CIN];
  __shared__ __align__(16) __bf16 Btile[2][SC * CIN];
  __shared__ float dbuf[32 * SC];

  const int t = threadIdx.x;
  const int wv = t >> 5, lane = t & 31;
  const int half = lane >> 4, lm = lane & 15;
  const int rowbase = blockIdx.x * 32;
  const int mt = wv >> 2, nt = wv & 3;

  {  // stage A tile (32 rows contiguous)
    const uint4* src = (const uint4*)(xb + (size_t)rowbase * CIN);
    uint4* dst = (uint4*)Atile;
#pragma unroll
    for (int i = 0; i < GA; ++i) dst[t + i * 256] = src[t + i * 256];
  }
  // prefetch first column group into buffer 0
#pragma unroll
  for (int i = 0; i < G; ++i) {
    int e = (t + i * 256) * 16;
    copy16_async((const char*)xb + e, (char*)Btile[0] + e);
  }

  float sqa[8];
#pragma unroll
  for (int v = 0; v < 8; ++v) sqa[v] = sq[rowbase + mt * 16 + half * 8 + v];

  float curV[4] = {__builtin_inff(), __builtin_inff(), __builtin_inff(),
                   __builtin_inff()};
  int curI[4] = {0, 0, 0, 0};

  int p = 0;
  for (int sc = 0; sc < N_PTS; sc += SC, p ^= 1) {
    if (sc + SC < N_PTS) {  // prefetch next group into the other buffer
#pragma unroll
      for (int i = 0; i < G; ++i) {
        int e = (t + i * 256) * 16;
        copy16_async((const char*)(xb + (size_t)(sc + SC) * CIN) + e,
                     (char*)Btile[p ^ 1] + e);
      }
      wait_async<G>();   // FIFO: current buffer's group has landed
    } else {
      wait_async<0>();
    }
    __syncthreads();

    // wave (mt,nt) computes its 16x16 Gram tile
    f32x8 acc = {};
#pragma unroll
    for (int kb = 0; kb < CIN; kb += 32) {
      bfx16 a = load_a(Atile + (mt * 16 + lm) * CIN, kb, half);
      bfx16 b = load_b(Btile[p] + (nt * 16 + lm) * CIN, kb, half);
      acc = wmma_bf16(a, b, acc);
    }
    const float sqc = sq[sc + nt * 16 + lm];
    const int   col = sc + nt * 16 + lm;
#pragma unroll
    for (int v = 0; v < 8; ++v) {
      int m = mt * 16 + half * 8 + v;
      float d = sqa[v] + sqc - 2.0f * acc[v];
      if (col == rowbase + m) d = __builtin_inff();  // exclude self
      dbuf[m * SC + nt * 16 + lm] = d;
    }
    __syncthreads();

    // selection: wave owns rows wv*4 .. wv*4+3; 2 batches of 32 per row
#pragma unroll
    for (int rr = 0; rr < 4; ++rr) {
      const int r = wv * 4 + rr;
#pragma unroll
      for (int bb = 0; bb < 2; ++bb) {
        float nv = dbuf[r * SC + bb * 32 + lane];
        int   ni = sc + bb * 32 + lane;
        float thr = __shfl(curV[rr], 31, 32);  // current 32nd-best
        if (__ballot(nv < thr))                // skip merge if nothing improves
          topk_merge(curV[rr], curI[rr], nv, ni);
      }
    }
    __syncthreads();
  }
#pragma unroll
  for (int rr = 0; rr < 4; ++rr)
    idx_out[(size_t)(rowbase + wv * 4 + rr) * KNN + lane] = curI[rr];
}

// ---- EdgeConv pass 1: h = relu(E @ W + b); per-node max/min over k --------
// plus atomically accumulated per-channel sum/sumsq for batch-norm.
// 2 nodes per WG -> M = 64 edge rows; N = 128 out channels; K = 2*CIN.
template <int CIN>
__global__ __launch_bounds__(256)
void edgeconv_pass1(const float* __restrict__ xin, const int* __restrict__ nbrs,
                    const __bf16* __restrict__ Wt, const float* __restrict__ bias,
                    float* __restrict__ hmax, float* __restrict__ hmin,
                    float* __restrict__ sums, float* __restrict__ sumsq) {
  constexpr int EC  = 2 * CIN;   // E width
  constexpr int NKB = EC / 64;   // 64-wide K chunks
  __shared__ __align__(16) __bf16 Ebuf[64 * 64];
  __shared__ __align__(16) __bf16 Wtc[128 * 64];
  __shared__ float hbuf[64 * 128];
  __shared__ int   nb[64];

  const int t = threadIdx.x;
  const int wv = t >> 5, lane = t & 31;
  const int half = lane >> 4, lm = lane & 15;
  const int i0 = blockIdx.x * 2;

  if (t < 64) nb[t] = nbrs[(size_t)(i0 + (t >> 5)) * KNN + (t & 31)];

  f32x8 acc[4] = {};

  for (int kb0 = 0; kb0 < NKB; ++kb0) {
    const int kb = kb0 * 64;
    __syncthreads();
    // async-stage W^T chunk (Wt is [128][EC] bf16): overlap with E gather
#pragma unroll
    for (int i = 0; i < 4; ++i) {
      int e = t + i * 256;            // 1024 x 16B chunks
      int c = e >> 3, kko = (e & 7) * 8;
      copy16_async(Wt + (size_t)c * EC + kb + kko, Wtc + c * 64 + kko);
    }
    // stage E chunk: row r = node*32+j, E = [x_i , x_j - x_i]
    for (int e = t; e < 64 * 64; e += 256) {
      int r = e >> 6, kk = e & 63;
      int k = kb + kk;
      int node = i0 + (r >> 5);
      float v;
      if (k < CIN) {
        v = xin[(size_t)node * CIN + k];
      } else {
        int k2 = k - CIN;
        v = xin[(size_t)nb[r] * CIN + k2] - xin[(size_t)node * CIN + k2];
      }
      Ebuf[e] = (__bf16)v;
    }
    wait_async<0>();
    __syncthreads();
#pragma unroll
    for (int tt = 0; tt < 4; ++tt) {      // 32 tiles / 8 waves = 4 each
      int tile = wv * 4 + tt;
      int mt = tile >> 3, nt = tile & 7;
#pragma unroll
      for (int ks = 0; ks < 64; ks += 32) {
        bfx16 a = load_a(Ebuf + (mt * 16 + lm) * 64, ks, half);
        bfx16 b = load_b(Wtc + (nt * 16 + lm) * 64, ks, half);
        acc[tt] = wmma_bf16(a, b, acc[tt]);
      }
    }
  }
  __syncthreads();
  // bias + relu into LDS (D layout: M = v + 8*half, N = lane%16)
#pragma unroll
  for (int tt = 0; tt < 4; ++tt) {
    int tile = wv * 4 + tt;
    int mt = tile >> 3, nt = tile & 7;
    int ch = nt * 16 + lm;
    float bv = bias[ch];
#pragma unroll
    for (int v = 0; v < 8; ++v) {
      int mrow = mt * 16 + half * 8 + v;
      float hv = acc[tt][v] + bv;
      hbuf[mrow * 128 + ch] = hv > 0.0f ? hv : 0.0f;
    }
  }
  __syncthreads();
  // per-(node,channel) max/min over k=32, plus channel sum/sumsq
  {
    int c = t & 127, nd = t >> 7;
    float s = 0.f, s2 = 0.f, mx = -__builtin_inff(), mn = __builtin_inff();
#pragma unroll 4
    for (int r = 0; r < 32; ++r) {
      float v = hbuf[(nd * 32 + r) * 128 + c];
      s += v; s2 += v * v;
      mx = v > mx ? v : mx;
      mn = v < mn ? v : mn;
    }
    hmax[(size_t)(i0 + nd) * 128 + c] = mx;
    hmin[(size_t)(i0 + nd) * 128 + c] = mn;
    atomicAdd(&sums[c], s);
    atomicAdd(&sumsq[c], s2);
  }
}

// ---- EdgeConv pass 2: BN affine + max-over-k via max/min trick ------------
template <bool RELU, bool WRITE_F32, bool WRITE_SQ>
__global__ __launch_bounds__(128)
void edgeconv_pass2(const float* __restrict__ hmax, const float* __restrict__ hmin,
                    const float* __restrict__ sums, const float* __restrict__ sumsq,
                    const float* __restrict__ g, const float* __restrict__ beta,
                    float* __restrict__ outf, __bf16* __restrict__ outb,
                    float* __restrict__ sqout) {
  __shared__ float red[128];
  const int c = threadIdx.x;
  const size_t row = blockIdx.x;
  const float cnt = (float)N_PTS * (float)KNN;
  float mu    = sums[c] / cnt;
  float var   = sumsq[c] / cnt - mu * mu;
  float scale = g[c] * rsqrtf(var + EPSV);
  float base  = scale >= 0.f ? hmax[row * 128 + c] : hmin[row * 128 + c];
  float o = scale * (base - mu) + beta[c];
  if (RELU) o = o > 0.f ? o : 0.f;
  if (WRITE_F32) outf[row * 128 + c] = o;
  outb[row * 128 + c] = (__bf16)o;
  if (WRITE_SQ) {
    red[c] = o * o;
    __syncthreads();
    for (int s = 64; s > 0; s >>= 1) {
      if (c < s) red[c] += red[c + s];
      __syncthreads();
    }
    if (c == 0) sqout[row] = red[0];
  }
}

// ---- Head MLP: relu(h2) @ lw1 -> relu -> @ lw2, all WMMA ------------------
__global__ __launch_bounds__(128)
void mlp_kernel(const __bf16* __restrict__ xin, const __bf16* __restrict__ w1t,
                const float* __restrict__ b1v, const __bf16* __restrict__ w2t,
                const float* __restrict__ b2v, float* __restrict__ out) {
  __shared__ __align__(16) __bf16 Abuf[16 * 128];
  __shared__ __align__(16) __bf16 W1s[64 * 128];
  __shared__ __align__(16) __bf16 W2s[64 * 64];
  __shared__ __align__(16) __bf16 T1[16 * 64];

  const int t = threadIdx.x;
  const int wv = t >> 5, lane = t & 31;
  const int half = lane >> 4, lm = lane & 15;
  const int rowbase = blockIdx.x * 16;

  {
    const uint4* src = (const uint4*)(xin + (size_t)rowbase * 128);
    uint4* dst = (uint4*)Abuf;
    for (int e = t; e < 16 * 128 / 8; e += 128) dst[e] = src[e];
    const uint4* s1 = (const uint4*)w1t;  uint4* d1 = (uint4*)W1s;
    for (int e = t; e < 64 * 128 / 8; e += 128) d1[e] = s1[e];
    const uint4* s2 = (const uint4*)w2t;  uint4* d2 = (uint4*)W2s;
    for (int e = t; e < 64 * 64 / 8; e += 128) d2[e] = s2[e];
  }
  __syncthreads();

  f32x8 acc = {};
#pragma unroll
  for (int kb = 0; kb < 128; kb += 32) {
    bfx16 a = load_a(Abuf + lm * 128, kb, half);
    bfx16 b = load_b(W1s + (wv * 16 + lm) * 128, kb, half);
    acc = wmma_bf16(a, b, acc);
  }
  {
    int n = wv * 16 + lm;
    float bv = b1v[n];
#pragma unroll
    for (int v = 0; v < 8; ++v) {
      int m = half * 8 + v;
      float h = acc[v] + bv;
      T1[m * 64 + n] = (__bf16)(h > 0.f ? h : 0.f);
    }
  }
  __syncthreads();

  f32x8 acc2 = {};
#pragma unroll
  for (int kb = 0; kb < 64; kb += 32) {
    bfx16 a = load_a(T1 + lm * 64, kb, half);
    bfx16 b = load_b(W2s + (wv * 16 + lm) * 64, kb, half);
    acc2 = wmma_bf16(a, b, acc2);
  }
  {
    int n = wv * 16 + lm;
    float bv = b2v[n];
#pragma unroll
    for (int v = 0; v < 8; ++v) {
      int m = half * 8 + v;
      out[(size_t)(rowbase + m) * NOUT + n] = acc2[v] + bv;
    }
  }
}

// ---- small helpers --------------------------------------------------------

__global__ __launch_bounds__(64)
void convert_x_kernel(const float* __restrict__ x, __bf16* __restrict__ xb,
                      float* __restrict__ sq) {
  __shared__ float red[64];
  const int c = threadIdx.x;
  const size_t row = blockIdx.x;
  float v = x[row * C0 + c];
  xb[row * C0 + c] = (__bf16)v;
  red[c] = v * v;
  __syncthreads();
  for (int s = 32; s > 0; s >>= 1) {
    if (c < s) red[c] += red[c + s];
    __syncthreads();
  }
  if (c == 0) sq[row] = red[0];
}

// Wt[c][k] = (bf16) W[k][c]   (W is [Kdim][Cdim] f32 row-major)
__global__ void transpose_bf16_kernel(const float* __restrict__ W,
                                      __bf16* __restrict__ Wt, int Kdim, int Cdim) {
  int e = blockIdx.x * blockDim.x + threadIdx.x;
  if (e >= Kdim * Cdim) return;
  int c = e / Kdim, k = e % Kdim;
  Wt[(size_t)c * Kdim + k] = (__bf16)W[(size_t)k * Cdim + c];
}

__global__ void zero_kernel(float* p, int n) {
  int e = blockIdx.x * blockDim.x + threadIdx.x;
  if (e < n) p[e] = 0.f;
}

// ---------------------------------------------------------------------------

extern "C" void kernel_launch(void* const* d_in, const int* in_sizes, int n_in,
                              void* d_out, int out_size, void* d_ws, size_t ws_size,
                              hipStream_t stream) {
  (void)in_sizes; (void)n_in; (void)out_size; (void)ws_size;
  const float* x   = (const float*)d_in[0];
  const float* W1  = (const float*)d_in[1];
  const float* b1  = (const float*)d_in[2];
  const float* g1  = (const float*)d_in[3];
  const float* be1 = (const float*)d_in[4];
  const float* W2  = (const float*)d_in[5];
  const float* b2  = (const float*)d_in[6];
  const float* g2  = (const float*)d_in[7];
  const float* be2 = (const float*)d_in[8];
  const float* lw1 = (const float*)d_in[9];
  const float* lb1 = (const float*)d_in[10];
  const float* lw2 = (const float*)d_in[11];
  const float* lb2 = (const float*)d_in[12];
  float* out = (float*)d_out;

  char* ws = (char*)d_ws;
  size_t off = 0;
  auto take = [&](size_t bytes) -> char* {
    char* p = ws + off;
    off = (off + bytes + 255) & ~(size_t)255;
    return p;
  };
  __bf16* xb    = (__bf16*)take((size_t)N_PTS * C0 * 2);
  float*  sq1   = (float*)take((size_t)N_PTS * 4);
  int*    idx1  = (int*)take((size_t)N_PTS * KNN * 4);
  __bf16* W1t   = (__bf16*)take((size_t)128 * 128 * 2);
  __bf16* W2t   = (__bf16*)take((size_t)128 * 256 * 2);
  __bf16* l1t   = (__bf16*)take((size_t)64 * 128 * 2);
  __bf16* l2t   = (__bf16*)take((size_t)64 * 64 * 2);
  float*  stats = (float*)take((size_t)512 * 4);  // sums1,sumsq1,sums2,sumsq2
  float*  hmax  = (float*)take((size_t)N_PTS * H * 4);
  float*  hmin  = (float*)take((size_t)N_PTS * H * 4);
  float*  h1f   = (float*)take((size_t)N_PTS * H * 4);
  __bf16* h1b   = (__bf16*)take((size_t)N_PTS * H * 2);
  float*  sq2   = (float*)take((size_t)N_PTS * 4);
  int*    idx2  = (int*)take((size_t)N_PTS * KNN * 4);
  __bf16* m2b   = (__bf16*)take((size_t)N_PTS * H * 2);

  convert_x_kernel<<<N_PTS, 64, 0, stream>>>(x, xb, sq1);
  transpose_bf16_kernel<<<(128 * 128 + 255) / 256, 256, 0, stream>>>(W1, W1t, 128, 128);
  transpose_bf16_kernel<<<(256 * 128 + 255) / 256, 256, 0, stream>>>(W2, W2t, 256, 128);
  transpose_bf16_kernel<<<(128 * 64 + 255) / 256, 256, 0, stream>>>(lw1, l1t, 128, 64);
  transpose_bf16_kernel<<<(64 * 64 + 255) / 256, 256, 0, stream>>>(lw2, l2t, 64, 64);
  zero_kernel<<<2, 256, 0, stream>>>(stats, 512);

  knn_kernel<64><<<N_PTS / 32, 256, 0, stream>>>(xb, sq1, idx1);
  edgeconv_pass1<64><<<N_PTS / 2, 256, 0, stream>>>(x, idx1, W1t, b1, hmax, hmin,
                                                    stats, stats + 128);
  edgeconv_pass2<false, true, true><<<N_PTS, 128, 0, stream>>>(
      hmax, hmin, stats, stats + 128, g1, be1, h1f, h1b, sq2);
  knn_kernel<128><<<N_PTS / 32, 256, 0, stream>>>(h1b, sq2, idx2);
  edgeconv_pass1<128><<<N_PTS / 2, 256, 0, stream>>>(h1f, idx2, W2t, b2, hmax, hmin,
                                                     stats + 256, stats + 384);
  edgeconv_pass2<true, false, false><<<N_PTS, 128, 0, stream>>>(
      hmax, hmin, stats + 256, stats + 384, g2, be2, nullptr, m2b, nullptr);
  mlp_kernel<<<N_PTS / 16, 128, 0, stream>>>(m2b, l1t, lb1, l2t, lb2, out);
}
